// Graph_29437705847196
// MI455X (gfx1250) — matrix operand order repaired and checked by reading, not compiled, not run on hardware
//
#include <hip/hip_runtime.h>
#include <hip/hip_bf16.h>

// ---------------------------------------------------------------------------
// CDNA5 (gfx1250) implementation of the stack-LSTM parser forward pass.
// All GEMMs run on v_wmma_f32_16x16x32_f16 (wave32). Weights are converted
// once to f16 [N][Kpad] layout so B-fragments are contiguous per lane.
// ---------------------------------------------------------------------------

typedef __attribute__((ext_vector_type(16))) _Float16 v16h;
typedef __attribute__((ext_vector_type(8)))  _Float16 v8h;
typedef __attribute__((ext_vector_type(8)))  float    v8f;

#define WMMA(a, b, c) __builtin_amdgcn_wmma_f32_16x16x32_f16(false, (a), false, (b), (short)0, (c), false, false)

__device__ __forceinline__ float sigf(float x) { return 1.0f / (1.0f + expf(-x)); }

// 16-bit A-matrix 16x32 fragment (ISA 7.12.2):
// lanes 0-15: M=lane,   halves 0-7 = K 0..7,  halves 8-15 = K 16..23
// lanes16-31: M=lane-16,halves 0-7 = K 8..15, halves 8-15 = K 24..31
__device__ __forceinline__ v16h load_afrag(const _Float16* p, int lane) {
  int off0 = (lane & 16) ? 8 : 0;
  v8h lo = *(const v8h*)(p + off0);
  v8h hi = *(const v8h*)(p + off0 + 16);
  v16h r;
#pragma unroll
  for (int i = 0; i < 8; i++) { r[i] = lo[i]; r[8 + i] = hi[i]; }
  return r;
}

__device__ __forceinline__ v16h load_afrag_f32(const float* p, int lane) {
  int off0 = (lane & 16) ? 8 : 0;
  v16h r;
#pragma unroll
  for (int i = 0; i < 8; i++) {
    r[i]     = (_Float16)p[off0 + i];
    r[8 + i] = (_Float16)p[off0 + 16 + i];
  }
  return r;
}

// B-matrix 32x16 fragment: lane n = col (lane&15); lanes 0-15 hold K 0..7,16..23,
// lanes 16-31 hold K 8..15,24..31 of weight row Wt[n][*] (Wt is [N][Kpad] f16).
__device__ __forceinline__ v16h load_bfrag(const _Float16* wt_row_k, int lane) {
  int off0 = (lane & 16) ? 8 : 0;
  v8h lo = *(const v8h*)(wt_row_k + off0);
  v8h hi = *(const v8h*)(wt_row_k + off0 + 16);
  v16h r;
#pragma unroll
  for (int i = 0; i < 8; i++) { r[i] = lo[i]; r[8 + i] = hi[i]; }
  return r;
}

// ---------------------------------------------------------------------------
// Weight prep: transpose fp32 [K][N] -> f16 [N][Kpad] (zero padded K tail)
// ---------------------------------------------------------------------------
__global__ void transpose_plain(const float* __restrict__ W, _Float16* __restrict__ Wt,
                                int K, int N, int Kpad) {
  int i = blockIdx.x * 256 + threadIdx.x;
  if (i < N * Kpad) {
    int n = i / Kpad, k = i % Kpad;
    Wt[i] = (_Float16)((k < K) ? W[(size_t)k * N + n] : 0.0f);
  }
}

// wd_W is [364][256]; embed A layout = [pos(64) | word(300) | pad(20)] -> Kpad 384
__global__ void transpose_wd(const float* __restrict__ W, _Float16* __restrict__ Wt) {
  int i = blockIdx.x * 256 + threadIdx.x;
  if (i < 256 * 384) {
    int n = i / 384, k = i % 384;
    float v = 0.0f;
    if (k < 64)       v = W[(size_t)(300 + k) * 256 + n];      // pos rows 300..363
    else if (k < 364) v = W[(size_t)(k - 64) * 256 + n];       // word rows 0..299
    Wt[i] = (_Float16)v;
  }
}

// fin_W is [1280][164] -> Wt [176][1280], rows >=164 zero
__global__ void transpose_fin(const float* __restrict__ W, _Float16* __restrict__ Wt) {
  int i = blockIdx.x * 256 + threadIdx.x;
  if (i < 176 * 1280) {
    int n = i / 1280, k = i % 1280;
    Wt[i] = (_Float16)((n < 164) ? W[(size_t)k * 164 + n] : 0.0f);
  }
}

// ---------------------------------------------------------------------------
// Embedding GEMM: out[row][256] = relu(concat(word,pos) @ wd_W + wd_b)
// A rows staged in LDS using [pos | word | pad] K layout. 64 rows per block.
// ---------------------------------------------------------------------------
__global__ void __launch_bounds__(256) embed_kernel(
    const int* __restrict__ wid, const int* __restrict__ pid, int nrows,
    const float* __restrict__ word_emb, const float* __restrict__ pos_emb,
    const _Float16* __restrict__ wt,   // [256][384]
    const float* __restrict__ bias,    // [256]
    _Float16* __restrict__ out)        // [nrows][256]
{
  __shared__ __align__(16) _Float16 sA[64 * 384];
  int row0 = blockIdx.x * 64;
  int tid = threadIdx.x;
  for (int e = tid; e < 64 * 384; e += 256) {
    int r = e / 384, k = e % 384;
    int row = row0 + r;
    float v = 0.0f;
    if (row < nrows) {
      if (k < 64)       v = pos_emb[(size_t)pid[row] * 64 + k];
      else if (k < 364) v = word_emb[(size_t)wid[row] * 300 + (k - 64)];
    }
    sA[e] = (_Float16)v;
  }
  __syncthreads();
  int wv = tid >> 5, lane = tid & 31;
  for (int p = wv; p < 64; p += 8) {       // 4 M-tiles x 16 N-tiles
    int mb = p >> 4, g = p & 15;
    int n = g * 16 + (lane & 15);
    v8f acc;
    float bi = bias[n];
#pragma unroll
    for (int r = 0; r < 8; r++) acc[r] = bi;
    for (int kb = 0; kb < 384; kb += 32) {
      v16h a = load_afrag(&sA[(mb * 16 + (lane & 15)) * 384 + kb], lane);
      v16h b = load_bfrag(wt + (size_t)n * 384 + kb, lane);
      acc = WMMA(a, b, acc);
    }
#pragma unroll
    for (int r = 0; r < 8; r++) {
      int m = r + ((lane & 16) ? 8 : 0);
      int row = row0 + mb * 16 + m;
      if (row < nrows) {
        float v = acc[r];
        out[(size_t)row * 256 + n] = (_Float16)(v > 0.0f ? v : 0.0f);
      }
    }
  }
}

// hist_e[b][a][64] = act_emb[id][.]  (f16)
__global__ void hist_kernel(const int* __restrict__ ids, const float* __restrict__ act_emb,
                            _Float16* __restrict__ out, int total) {
  int i = blockIdx.x * 256 + threadIdx.x;
  if (i < total) {
    int row = i >> 6, k = i & 63;
    out[i] = (_Float16)act_emb[(size_t)ids[row] * 64 + k];
  }
}

// ---------------------------------------------------------------------------
// Tree-LSTM: single workgroup, 128 sequential steps.
//  G1: [x|hj] @ W[:, :768] -> i,j,o    G2: [x|h_k] @ W[:,768:] -> f (256 rows)
// ---------------------------------------------------------------------------
__global__ void __launch_bounds__(512) tree_kernel(
    const _Float16* __restrict__ tree_e,   // [64][128][256] f16
    const int* __restrict__ children,      // [64][128][4]
    const _Float16* __restrict__ wt,       // [1024][512] f16
    const float* __restrict__ bias,        // [1024]
    float* __restrict__ H, float* __restrict__ C,   // [129][64][256]
    float* __restrict__ z1,                // [64][768] scratch
    float* __restrict__ fbuf,              // [256][256] scratch
    _Float16* __restrict__ hjbuf)          // [64][256] f16 scratch
{
  int tid = threadIdx.x;
  for (int e = tid; e < 64 * 256; e += 512) { H[e] = 0.0f; C[e] = 0.0f; }
  __threadfence();
  __syncthreads();
  int wv = tid >> 5, lane = tid & 31;

  for (int i = 0; i < 128; i++) {
    // phase 0: hj = sum_k H[child_k]
    for (int e = tid; e < 64 * 256; e += 512) {
      int b = e >> 8, u = e & 255;
      const int* ch = children + (b * 128 + i) * 4;
      float s = 0.0f;
#pragma unroll
      for (int k = 0; k < 4; k++) s += H[((size_t)ch[k] * 64 + b) * 256 + u];
      hjbuf[e] = (_Float16)s;
    }
    __threadfence();
    __syncthreads();

    // phase 1: z1(i,j,o) = [x|hj] @ W[:, :768]
    for (int q = 0; q < 4; q++) {
      int p = wv * 4 + q, mb = p >> 4, g = p & 15;
      int mrow = mb * 16 + (lane & 15);
      const _Float16* px = tree_e + ((size_t)mrow * 128 + i) * 256;
      v8f acc[3];
#pragma unroll
      for (int gate = 0; gate < 3; gate++) {
        float bi = bias[gate * 256 + g * 16 + (lane & 15)];
#pragma unroll
        for (int r = 0; r < 8; r++) acc[gate][r] = bi;
      }
      for (int kb = 0; kb < 512; kb += 32) {
        v16h a = (kb < 256) ? load_afrag(px + kb, lane)
                            : load_afrag(hjbuf + mrow * 256 + (kb - 256), lane);
#pragma unroll
        for (int gate = 0; gate < 3; gate++) {
          int n = gate * 256 + g * 16 + (lane & 15);
          v16h bf = load_bfrag(wt + (size_t)n * 512 + kb, lane);
          acc[gate] = WMMA(a, bf, acc[gate]);
        }
      }
#pragma unroll
      for (int gate = 0; gate < 3; gate++)
#pragma unroll
        for (int r = 0; r < 8; r++) {
          int m = r + ((lane & 16) ? 8 : 0);
          int b = mb * 16 + m;
          z1[b * 768 + gate * 256 + g * 16 + (lane & 15)] = acc[gate][r];
        }
    }

    // phase 2: f[b,k] = [x|h_k] @ W[:, 768:]   (256 rows)
    for (int p2 = wv; p2 < 256; p2 += 16) {
      int m2 = p2 >> 4, g = p2 & 15;
      int rr = m2 * 16 + (lane & 15);
      int b = rr >> 2, k = rr & 3;
      int node = children[(b * 128 + i) * 4 + k];
      const _Float16* px = tree_e + ((size_t)b * 128 + i) * 256;
      const float* ph = H + ((size_t)node * 64 + b) * 256;
      v8f acc;
      {
        float bi = bias[768 + g * 16 + (lane & 15)];
#pragma unroll
        for (int r = 0; r < 8; r++) acc[r] = bi;
      }
      int n = 768 + g * 16 + (lane & 15);
      for (int kb = 0; kb < 512; kb += 32) {
        v16h a = (kb < 256) ? load_afrag(px + kb, lane)
                            : load_afrag_f32(ph + (kb - 256), lane);
        v16h bf = load_bfrag(wt + (size_t)n * 512 + kb, lane);
        acc = WMMA(a, bf, acc);
      }
#pragma unroll
      for (int r = 0; r < 8; r++) {
        int m = r + ((lane & 16) ? 8 : 0);
        fbuf[(m2 * 16 + m) * 256 + g * 16 + (lane & 15)] = acc[r];
      }
    }
    __threadfence();
    __syncthreads();

    // phase 3: combine
    for (int e = tid; e < 64 * 256; e += 512) {
      int b = e >> 8, u = e & 255;
      const int* ch = children + (b * 128 + i) * 4;
      float iv = z1[b * 768 + u];
      float jv = z1[b * 768 + 256 + u];
      float ov = z1[b * 768 + 512 + u];
      float nc = sigf(iv) * tanhf(jv);
#pragma unroll
      for (int k = 0; k < 4; k++) {
        float fv = fbuf[(b * 4 + k) * 256 + u];
        float cv = C[((size_t)ch[k] * 64 + b) * 256 + u];
        nc += cv * sigf(fv + 1.0f);
      }
      float nh = tanhf(nc) * sigf(ov);
      H[((size_t)(i + 1) * 64 + b) * 256 + u] = nh;
      C[((size_t)(i + 1) * 64 + b) * 256 + u] = nc;
    }
    __threadfence();
    __syncthreads();
  }
}

// ---------------------------------------------------------------------------
// Generic LSTM chain: one workgroup (512 threads = 16 waves), sequential steps.
// h (f16) in LDS; c carried in wave-private accumulator registers.
// ---------------------------------------------------------------------------
struct ChainDesc {
  const _Float16* xA;
  const _Float16* xB;
  const _Float16* Wt;       // [1024][Kpad] f16
  const float* bias;        // [1024]
  const int* len;           // [64]
  _Float16* out;            // [64][Ln][256] f16
  float* hfin;              // nullable [64][256] fp32
  int DinA, DinB, revA, revB, Ln, Kpad;
};

__device__ void run_chain(const ChainDesc& d) {
  __shared__ __align__(16) _Float16 sh[64 * 256];
  __shared__ float sbias[1024];
  __shared__ int slen[64];
  int tid = threadIdx.x;
  for (int e = tid; e < 64 * 256; e += 512) sh[e] = (_Float16)0.0f;
  for (int e = tid; e < 1024; e += 512) sbias[e] = d.bias[e];
  if (tid < 64) slen[tid] = d.len[tid];
  __syncthreads();

  int wv = tid >> 5, lane = tid & 31;
  int ml = lane & 15;
  int KpA = d.DinA, KpAB = d.DinA + d.DinB;

  v8f cc[4];
#pragma unroll
  for (int q = 0; q < 4; q++)
#pragma unroll
    for (int r = 0; r < 8; r++) cc[q][r] = 0.0f;

  for (int t = 0; t < d.Ln; t++) {
    v8f acc[4][4];
#pragma unroll
    for (int q = 0; q < 4; q++) {
      int p = wv * 4 + q, mb = p >> 4, g = p & 15;
      int b = mb * 16 + ml;
      int lb = slen[b];
      int tA = (d.revA && t < lb) ? (lb - 1 - t) : t;
      const _Float16* pA = d.xA + ((size_t)b * d.Ln + tA) * d.DinA;
      const _Float16* pB = nullptr;
      if (d.xB) {
        int tB = (d.revB && t < lb) ? (lb - 1 - t) : t;
        pB = d.xB + ((size_t)b * d.Ln + tB) * d.DinB;
      }
#pragma unroll
      for (int gate = 0; gate < 4; gate++) {
        float bi = sbias[gate * 256 + g * 16 + ml];
#pragma unroll
        for (int r = 0; r < 8; r++) acc[q][gate][r] = bi;
      }
      for (int kb = 0; kb < d.Kpad; kb += 32) {
        v16h a;
        if (kb < KpA)       a = load_afrag(pA + kb, lane);
        else if (kb < KpAB) a = load_afrag(pB + (kb - KpA), lane);
        else                a = load_afrag(sh + b * 256 + (kb - KpAB), lane);
#pragma unroll
        for (int gate = 0; gate < 4; gate++) {
          int n = gate * 256 + g * 16 + ml;
          v16h bf = load_bfrag(d.Wt + (size_t)n * d.Kpad + kb, lane);
          acc[q][gate] = WMMA(a, bf, acc[q][gate]);
        }
      }
    }
    __syncthreads();   // all LDS h reads done before rewriting h
#pragma unroll
    for (int q = 0; q < 4; q++) {
      int p = wv * 4 + q, mb = p >> 4, g = p & 15;
#pragma unroll
      for (int r = 0; r < 8; r++) {
        int m = r + ((lane & 16) ? 8 : 0);
        int b = mb * 16 + m;
        int u = g * 16 + ml;
        float iv = acc[q][0][r], jv = acc[q][1][r], fv = acc[q][2][r], ov = acc[q][3][r];
        float ncv = cc[q][r] * sigf(fv + 1.0f) + sigf(iv) * tanhf(jv);
        float nhv = tanhf(ncv) * sigf(ov);
        bool mk = (t < slen[b]);
        _Float16 hv = (_Float16)nhv;
        d.out[((size_t)b * d.Ln + t) * 256 + u] = mk ? hv : (_Float16)0.0f;
        if (mk) { cc[q][r] = ncv; sh[b * 256 + u] = hv; }
      }
    }
    __syncthreads();   // h writes visible before next step's reads
  }

  if (d.hfin) {
    for (int e = tid; e < 64 * 256; e += 512) d.hfin[e] = (float)sh[e];
  }
}

__global__ void __launch_bounds__(512) chains_kernel(
    ChainDesc c0, ChainDesc c1, ChainDesc c2, ChainDesc c3, ChainDesc c4,
    const int* __restrict__ stack_order, const float* __restrict__ H,
    _Float16* __restrict__ xs, int do_gather)
{
  if (do_gather && blockIdx.x == 0) {
    // stack_embedded = H[stack_order[b][s], b]  -> xs f16 [b][s][256]
    for (int e = threadIdx.x; e < 64 * 64 * 256; e += 512) {
      int b = e / (64 * 256);
      int rest = e % (64 * 256);
      int s = rest >> 8, u = rest & 255;
      int node = stack_order[b * 64 + s];
      xs[e] = (_Float16)H[((size_t)node * 64 + b) * 256 + u];
    }
    __threadfence();
    __syncthreads();
  }
  ChainDesc d = (blockIdx.x == 0) ? c0 : (blockIdx.x == 1) ? c1 :
                (blockIdx.x == 2) ? c2 : (blockIdx.x == 3) ? c3 : c4;
  run_chain(d);
}

// ---------------------------------------------------------------------------
// Final: build feat[64][1280] then out = relu(feat @ fin_W + fin_b) (64x164)
// ---------------------------------------------------------------------------
__global__ void __launch_bounds__(256) final_kernel(
    const float* __restrict__ stack_h, const float* __restrict__ fw1_h,
    const _Float16* __restrict__ fw1_o, const _Float16* __restrict__ bw1_o,
    const float* __restrict__ act_h, const float* __restrict__ dq_h,
    const int* __restrict__ buff_top, const int* __restrict__ tok_len,
    const _Float16* __restrict__ wt_fin,   // [176][1280]
    const float* __restrict__ fin_b,       // [164]
    _Float16* __restrict__ feat,           // [64][1280]
    float* __restrict__ out)               // [64][164]
{
  int tid = threadIdx.x;
  for (int e = tid; e < 64 * 1280; e += 256) {
    int b = e / 1280, u = e % 1280;
    float v;
    if (u < 256) {
      v = stack_h[b * 256 + u];
    } else if (u < 512) {
      int uu = u - 256;
      float ftop = (float)fw1_o[((size_t)b * 128 + buff_top[b]) * 256 + uu];
      v = fw1_h[b * 256 + uu] - ftop;
    } else if (u < 768) {
      int uu = u - 512;
      int lb = tok_len[b], bt = buff_top[b];
      int idx_top = (bt < lb) ? (lb - 1 - bt) : bt;   // bw1_o is in reversed domain
      float btop = (float)bw1_o[((size_t)b * 128 + idx_top) * 256 + uu];
      float bbot = (float)bw1_o[((size_t)b * 128 + 0) * 256 + uu]; // rev index of (len-1)
      v = btop - bbot;
    } else if (u < 1024) {
      v = act_h[b * 256 + (u - 768)];
    } else {
      v = dq_h[b * 256 + (u - 1024)];
    }
    feat[e] = (_Float16)v;
  }
  __threadfence();
  __syncthreads();

  int wv = tid >> 5, lane = tid & 31;
  for (int p = wv; p < 44; p += 8) {     // 4 M-tiles x 11 N-tiles (176 cols)
    int mb = p / 11, nt = p % 11;
    int n = nt * 16 + (lane & 15);
    v8f acc;
    float bi = (n < 164) ? fin_b[n] : 0.0f;
#pragma unroll
    for (int r = 0; r < 8; r++) acc[r] = bi;
    for (int kb = 0; kb < 1280; kb += 32) {
      v16h a = load_afrag(feat + (size_t)(mb * 16 + (lane & 15)) * 1280 + kb, lane);
      v16h bf = load_bfrag(wt_fin + (size_t)n * 1280 + kb, lane);
      acc = WMMA(a, bf, acc);
    }
    if (n < 164) {
#pragma unroll
      for (int r = 0; r < 8; r++) {
        int m = r + ((lane & 16) ? 8 : 0);
        int b = mb * 16 + m;
        float v = acc[r];
        out[b * 164 + n] = v > 0.0f ? v : 0.0f;
      }
    }
  }
}

// ---------------------------------------------------------------------------
// Host launcher
// ---------------------------------------------------------------------------
extern "C" void kernel_launch(void* const* d_in, const int* in_sizes, int n_in,
                              void* d_out, int out_size, void* d_ws, size_t ws_size,
                              hipStream_t stream) {
  (void)in_sizes; (void)n_in; (void)out_size; (void)ws_size;

  const int* tree_word_id   = (const int*)d_in[0];
  const int* tree_pos_id    = (const int*)d_in[1];
  const int* token_word_id  = (const int*)d_in[2];
  const int* token_pos_id   = (const int*)d_in[3];
  const int* hist_action_id = (const int*)d_in[4];
  const int* buff_top_id    = (const int*)d_in[5];
  const int* deque_word_id  = (const int*)d_in[6];
  const int* deque_pos_id   = (const int*)d_in[7];
  const int* deque_length   = (const int*)d_in[8];
  const int* children_order = (const int*)d_in[9];
  const int* stack_order    = (const int*)d_in[10];
  const int* stack_length   = (const int*)d_in[11];
  const int* token_length   = (const int*)d_in[12];
  const int* hist_length    = (const int*)d_in[13];
  const float* word_emb = (const float*)d_in[14];
  const float* pos_emb  = (const float*)d_in[15];
  const float* act_emb  = (const float*)d_in[16];
  const float* wd_W   = (const float*)d_in[17];
  const float* wd_b   = (const float*)d_in[18];
  const float* tree_W = (const float*)d_in[19];
  const float* tree_b = (const float*)d_in[20];
  const float* stack0_W = (const float*)d_in[21];
  const float* stack0_b = (const float*)d_in[22];
  const float* stack1_W = (const float*)d_in[23];
  const float* stack1_b = (const float*)d_in[24];
  const float* fw0_W = (const float*)d_in[25];
  const float* fw0_b = (const float*)d_in[26];
  const float* bw0_W = (const float*)d_in[27];
  const float* bw0_b = (const float*)d_in[28];
  const float* fw1_W = (const float*)d_in[29];
  const float* fw1_b = (const float*)d_in[30];
  const float* bw1_W = (const float*)d_in[31];
  const float* bw1_b = (const float*)d_in[32];
  const float* act0_W = (const float*)d_in[33];
  const float* act0_b = (const float*)d_in[34];
  const float* act1_W = (const float*)d_in[35];
  const float* act1_b = (const float*)d_in[36];
  const float* dq0_W = (const float*)d_in[37];
  const float* dq0_b = (const float*)d_in[38];
  const float* dq1_W = (const float*)d_in[39];
  const float* dq1_b = (const float*)d_in[40];
  const float* fin_W = (const float*)d_in[41];
  const float* fin_b = (const float*)d_in[42];

  char* wp = (char*)d_ws;
  auto alloc = [&](size_t bytes) -> char* {
    char* r = wp;
    wp += (bytes + 255) & ~(size_t)255;
    return r;
  };

  _Float16* wt_wd     = (_Float16*)alloc((size_t)256 * 384 * 2);
  _Float16* wt_tree   = (_Float16*)alloc((size_t)1024 * 512 * 2);
  _Float16* wt_stack0 = (_Float16*)alloc((size_t)1024 * 512 * 2);
  _Float16* wt_stack1 = (_Float16*)alloc((size_t)1024 * 512 * 2);
  _Float16* wt_fw0    = (_Float16*)alloc((size_t)1024 * 512 * 2);
  _Float16* wt_bw0    = (_Float16*)alloc((size_t)1024 * 512 * 2);
  _Float16* wt_fw1    = (_Float16*)alloc((size_t)1024 * 768 * 2);
  _Float16* wt_bw1    = (_Float16*)alloc((size_t)1024 * 768 * 2);
  _Float16* wt_act0   = (_Float16*)alloc((size_t)1024 * 320 * 2);
  _Float16* wt_act1   = (_Float16*)alloc((size_t)1024 * 512 * 2);
  _Float16* wt_dq0    = (_Float16*)alloc((size_t)1024 * 512 * 2);
  _Float16* wt_dq1    = (_Float16*)alloc((size_t)1024 * 512 * 2);
  _Float16* wt_fin    = (_Float16*)alloc((size_t)176 * 1280 * 2);

  _Float16* tree_e  = (_Float16*)alloc((size_t)64 * 128 * 256 * 2);
  _Float16* token_e = (_Float16*)alloc((size_t)64 * 128 * 256 * 2);
  _Float16* deque_e = (_Float16*)alloc((size_t)64 * 32 * 256 * 2);
  _Float16* hist_e  = (_Float16*)alloc((size_t)64 * 64 * 64 * 2);

  float* H = (float*)alloc((size_t)129 * 64 * 256 * 4);
  float* C = (float*)alloc((size_t)129 * 64 * 256 * 4);
  float* z1   = (float*)alloc((size_t)64 * 768 * 4);
  float* fbuf = (float*)alloc((size_t)256 * 256 * 4);
  _Float16* hjbuf = (_Float16*)alloc((size_t)64 * 256 * 2);

  _Float16* xs    = (_Float16*)alloc((size_t)64 * 64 * 256 * 2);
  _Float16* o0    = (_Float16*)alloc((size_t)64 * 64 * 256 * 2);
  _Float16* fw0_o = (_Float16*)alloc((size_t)64 * 128 * 256 * 2);
  _Float16* bw0_o = (_Float16*)alloc((size_t)64 * 128 * 256 * 2);
  _Float16* fw1_o = (_Float16*)alloc((size_t)64 * 128 * 256 * 2);
  _Float16* bw1_o = (_Float16*)alloc((size_t)64 * 128 * 256 * 2);
  _Float16* a0    = (_Float16*)alloc((size_t)64 * 64 * 256 * 2);
  _Float16* d0    = (_Float16*)alloc((size_t)64 * 32 * 256 * 2);
  _Float16* o1d   = (_Float16*)alloc((size_t)64 * 64 * 256 * 2);
  _Float16* a1d   = (_Float16*)alloc((size_t)64 * 64 * 256 * 2);
  _Float16* d1d   = (_Float16*)alloc((size_t)64 * 32 * 256 * 2);

  float* stack_h = (float*)alloc((size_t)64 * 256 * 4);
  float* fw1_h   = (float*)alloc((size_t)64 * 256 * 4);
  float* act_h   = (float*)alloc((size_t)64 * 256 * 4);
  float* dq_h    = (float*)alloc((size_t)64 * 256 * 4);
  _Float16* feat = (_Float16*)alloc((size_t)64 * 1280 * 2);

  // --- weight prep ---
  auto tp = [&](const float* W, _Float16* Wt, int K, int N, int Kpad) {
    int total = N * Kpad;
    transpose_plain<<<(total + 255) / 256, 256, 0, stream>>>(W, Wt, K, N, Kpad);
  };
  transpose_wd<<<(256 * 384 + 255) / 256, 256, 0, stream>>>(wd_W, wt_wd);
  tp(tree_W,   wt_tree,   512, 1024, 512);
  tp(stack0_W, wt_stack0, 512, 1024, 512);
  tp(stack1_W, wt_stack1, 512, 1024, 512);
  tp(fw0_W,    wt_fw0,    512, 1024, 512);
  tp(bw0_W,    wt_bw0,    512, 1024, 512);
  tp(fw1_W,    wt_fw1,    768, 1024, 768);
  tp(bw1_W,    wt_bw1,    768, 1024, 768);
  tp(act0_W,   wt_act0,   320, 1024, 320);
  tp(act1_W,   wt_act1,   512, 1024, 512);
  tp(dq0_W,    wt_dq0,    512, 1024, 512);
  tp(dq1_W,    wt_dq1,    512, 1024, 512);
  transpose_fin<<<(176 * 1280 + 255) / 256, 256, 0, stream>>>(fin_W, wt_fin);

  // --- embeddings ---
  embed_kernel<<<(64 * 128) / 64, 256, 0, stream>>>(tree_word_id, tree_pos_id, 64 * 128,
                                                    word_emb, pos_emb, wt_wd, wd_b, tree_e);
  embed_kernel<<<(64 * 128) / 64, 256, 0, stream>>>(token_word_id, token_pos_id, 64 * 128,
                                                    word_emb, pos_emb, wt_wd, wd_b, token_e);
  embed_kernel<<<(64 * 32) / 64, 256, 0, stream>>>(deque_word_id, deque_pos_id, 64 * 32,
                                                   word_emb, pos_emb, wt_wd, wd_b, deque_e);
  hist_kernel<<<(64 * 64 * 64 + 255) / 256, 256, 0, stream>>>(hist_action_id, act_emb,
                                                              hist_e, 64 * 64 * 64);

  // --- tree LSTM ---
  tree_kernel<<<1, 512, 0, stream>>>(tree_e, children_order, wt_tree, tree_b,
                                     H, C, z1, fbuf, hjbuf);

  auto mk = [](const _Float16* xA, int DinA, int revA,
               const _Float16* xB, int DinB, int revB,
               int Ln, const int* len, const _Float16* Wt, int Kpad,
               const float* bias, _Float16* out, float* hfin) {
    ChainDesc d;
    d.xA = xA; d.xB = xB; d.Wt = Wt; d.bias = bias; d.len = len;
    d.out = out; d.hfin = hfin;
    d.DinA = DinA; d.DinB = DinB; d.revA = revA; d.revB = revB;
    d.Ln = Ln; d.Kpad = Kpad;
    return d;
  };

  // --- level 1: stack0, fw0, bw0, act0, dq0 ---
  chains_kernel<<<5, 512, 0, stream>>>(
      mk(xs,      256, 0, nullptr, 0, 0,  64, stack_length, wt_stack0, 512, stack0_b, o0,    nullptr),
      mk(token_e, 256, 0, nullptr, 0, 0, 128, token_length, wt_fw0,    512, fw0_b,    fw0_o, nullptr),
      mk(token_e, 256, 1, nullptr, 0, 0, 128, token_length, wt_bw0,    512, bw0_b,    bw0_o, nullptr),
      mk(hist_e,   64, 0, nullptr, 0, 0,  64, hist_length,  wt_act0,   320, act0_b,   a0,    nullptr),
      mk(deque_e, 256, 0, nullptr, 0, 0,  32, deque_length, wt_dq0,    512, dq0_b,    d0,    nullptr),
      stack_order, H, xs, 1);

  // --- level 2: stack1, fw1, bw1, act1, dq1 ---
  chains_kernel<<<5, 512, 0, stream>>>(
      mk(o0,    256, 0, nullptr, 0, 0,  64, stack_length, wt_stack1, 512, stack1_b, o1d,   stack_h),
      mk(fw0_o, 256, 0, bw0_o, 256, 1, 128, token_length, wt_fw1,    768, fw1_b,    fw1_o, fw1_h),
      mk(fw0_o, 256, 1, bw0_o, 256, 0, 128, token_length, wt_bw1,    768, bw1_b,    bw1_o, nullptr),
      mk(a0,    256, 0, nullptr, 0, 0,  64, hist_length,  wt_act1,   512, act1_b,   a1d,   act_h),
      mk(d0,    256, 0, nullptr, 0, 0,  32, deque_length, wt_dq1,    512, dq1_b,    d1d,   dq_h),
      stack_order, H, xs, 0);

  // --- final projection ---
  final_kernel<<<1, 256, 0, stream>>>(stack_h, fw1_h, fw1_o, bw1_o, act_h, dq_h,
                                      buff_top_id, token_length, wt_fin, fin_b,
                                      feat, (float*)d_out);
}
